// GroupQueryAttention_59416577573503
// MI455X (gfx1250) — matrix operand order repaired and checked by reading, not compiled, not run on hardware
//
#include <hip/hip_runtime.h>
#include <hip/hip_bf16.h>

typedef __attribute__((ext_vector_type(16))) __bf16 v16bf;
typedef __attribute__((ext_vector_type(8)))  __bf16 v8bf;
typedef __attribute__((ext_vector_type(8)))  float  v8f;

#define EMBED    2048
#define HEADS    32
#define GROUPS   8
#define HEAD_DIM 64
#define BATCH    4
#define SEQ      1024
#define ROWS     (BATCH * SEQ)        // 4096
#define KV       (GROUPS * HEAD_DIM)  // 512

// ---------------------------------------------------------------------------
// WMMA helper: D = A(16x32 bf16) * B(32x16 bf16) + C(16x16 f32)
// ---------------------------------------------------------------------------
__device__ __forceinline__ v8f wmma_bf16(v16bf a, v16bf b, v8f c) {
  return __builtin_amdgcn_wmma_f32_16x16x32_bf16(
      /*neg_a=*/false, a, /*neg_b=*/false, b,
      /*c_mod=*/(short)0, c, /*reuse_a=*/false, /*reuse_b=*/false);
}

__device__ __forceinline__ v16bf combine16(v8bf lo, v8bf hi) {
  v16bf r;
#pragma unroll
  for (int i = 0; i < 8; ++i) { r[i] = lo[i]; r[i + 8] = hi[i]; }
  return r;
}

// 8 contiguous f32 -> 8 bf16
__device__ __forceinline__ v8bf cvt8(const float* p) {
  float4 x = *(const float4*)p;
  float4 y = *(const float4*)(p + 4);
  v8bf r;
  r[0] = (__bf16)x.x; r[1] = (__bf16)x.y; r[2] = (__bf16)x.z; r[3] = (__bf16)x.w;
  r[4] = (__bf16)y.x; r[5] = (__bf16)y.y; r[6] = (__bf16)y.z; r[7] = (__bf16)y.w;
  return r;
}

// A-fragment (16x32 bf16): lane m=l%16, half=l/16.
// elem e -> k = half*16 + e
__device__ __forceinline__ v16bf loadA_f32(const float* rowp, int half) {
  return combine16(cvt8(rowp + half * 16), cvt8(rowp + half * 16 + 8));
}
__device__ __forceinline__ v16bf loadA_bf16(const __bf16* rowp, int half) {
  v8bf lo = *(const v8bf*)(rowp + half * 16);
  v8bf hi = *(const v8bf*)(rowp + half * 16 + 8);
  return combine16(lo, hi);
}

// B-fragment (32x16 bf16): lane n=l%16, half=l/16; elem e -> k = half*16 + e
__device__ __forceinline__ v16bf loadB_f32(const float* rowp, int half) {
  const float* p = rowp + half * 16;
  float4 x0 = *(const float4*)(p);
  float4 x1 = *(const float4*)(p + 4);
  float4 x2 = *(const float4*)(p + 8);
  float4 x3 = *(const float4*)(p + 12);
  v16bf r;
  r[0]  = (__bf16)x0.x; r[1]  = (__bf16)x0.y; r[2]  = (__bf16)x0.z; r[3]  = (__bf16)x0.w;
  r[4]  = (__bf16)x1.x; r[5]  = (__bf16)x1.y; r[6]  = (__bf16)x1.z; r[7]  = (__bf16)x1.w;
  r[8]  = (__bf16)x2.x; r[9]  = (__bf16)x2.y; r[10] = (__bf16)x2.z; r[11] = (__bf16)x2.w;
  r[12] = (__bf16)x3.x; r[13] = (__bf16)x3.y; r[14] = (__bf16)x3.z; r[15] = (__bf16)x3.w;
  return r;
}

// ---------------------------------------------------------------------------
// GEMM: C[M,N](bf16) = X[M,2048](f32) * W[N,2048](f32)^T
// wave tile 64x64, block = 4 waves = 64m x 256n, grid (M/64, N/256)
// ---------------------------------------------------------------------------
__global__ __launch_bounds__(128) void gemm_f32_bf16(
    const float* __restrict__ X, const float* __restrict__ W,
    __bf16* __restrict__ C, int N) {
  const int lane = threadIdx.x & 31, wave = threadIdx.x >> 5;
  const int l16 = lane & 15, half = lane >> 4;
  const int m0 = blockIdx.x * 64;
  const int n0 = blockIdx.y * 256 + wave * 64;

  v8f acc[4][4];
#pragma unroll
  for (int i = 0; i < 4; ++i)
#pragma unroll
    for (int j = 0; j < 4; ++j) acc[i][j] = (v8f){0,0,0,0,0,0,0,0};

  for (int kc = 0; kc < EMBED; kc += 32) {
    // prefetch next k-chunk (speculative; emits global_prefetch_b8)
    __builtin_prefetch(X + (size_t)(m0 + l16) * EMBED + kc + 64, 0, 0);
    __builtin_prefetch(W + (size_t)(n0 + l16) * EMBED + kc + 64, 0, 0);
    v16bf aF[4], bF[4];
#pragma unroll
    for (int mt = 0; mt < 4; ++mt)
      aF[mt] = loadA_f32(X + (size_t)(m0 + mt * 16 + l16) * EMBED + kc, half);
#pragma unroll
    for (int nt = 0; nt < 4; ++nt)
      bF[nt] = loadB_f32(W + (size_t)(n0 + nt * 16 + l16) * EMBED + kc, half);
#pragma unroll
    for (int mt = 0; mt < 4; ++mt)
#pragma unroll
      for (int nt = 0; nt < 4; ++nt)
        acc[mt][nt] = wmma_bf16(aF[mt], bF[nt], acc[mt][nt]);
  }

#pragma unroll
  for (int mt = 0; mt < 4; ++mt)
#pragma unroll
    for (int nt = 0; nt < 4; ++nt)
#pragma unroll
      for (int r = 0; r < 8; ++r) {
        int m = m0 + mt * 16 + r + 8 * half;
        int n = n0 + nt * 16 + l16;
        C[(size_t)m * N + n] = (__bf16)acc[mt][nt][r];
      }
}

// ---------------------------------------------------------------------------
// V projection with TRANSPOSED output: Vt[b][n][s] = (v @ Wv^T)[b*S+s][n]
// Vt layout: [(b*KV + n) * SEQ + s], s-contiguous -> A.V B-fragments in
// attention become single contiguous 32B loads.
// ---------------------------------------------------------------------------
__global__ __launch_bounds__(128) void gemm_f32_bf16_vT(
    const float* __restrict__ X, const float* __restrict__ W,
    __bf16* __restrict__ Vt) {
  const int lane = threadIdx.x & 31, wave = threadIdx.x >> 5;
  const int l16 = lane & 15, half = lane >> 4;
  const int m0 = blockIdx.x * 64;
  const int n0 = blockIdx.y * 256 + wave * 64;

  v8f acc[4][4];
#pragma unroll
  for (int i = 0; i < 4; ++i)
#pragma unroll
    for (int j = 0; j < 4; ++j) acc[i][j] = (v8f){0,0,0,0,0,0,0,0};

  for (int kc = 0; kc < EMBED; kc += 32) {
    __builtin_prefetch(X + (size_t)(m0 + l16) * EMBED + kc + 64, 0, 0);
    __builtin_prefetch(W + (size_t)(n0 + l16) * EMBED + kc + 64, 0, 0);
    v16bf aF[4], bF[4];
#pragma unroll
    for (int mt = 0; mt < 4; ++mt)
      aF[mt] = loadA_f32(X + (size_t)(m0 + mt * 16 + l16) * EMBED + kc, half);
#pragma unroll
    for (int nt = 0; nt < 4; ++nt)
      bF[nt] = loadB_f32(W + (size_t)(n0 + nt * 16 + l16) * EMBED + kc, half);
#pragma unroll
    for (int mt = 0; mt < 4; ++mt)
#pragma unroll
      for (int nt = 0; nt < 4; ++nt)
        acc[mt][nt] = wmma_bf16(aF[mt], bF[nt], acc[mt][nt]);
  }

#pragma unroll
  for (int mt = 0; mt < 4; ++mt)
#pragma unroll
    for (int nt = 0; nt < 4; ++nt)
#pragma unroll
      for (int r = 0; r < 8; ++r) {
        int m = m0 + mt * 16 + r + 8 * half;   // m = b*SEQ + s
        int n = n0 + nt * 16 + l16;            // n = g*64 + d
        int bb = m >> 10;                      // SEQ = 1024
        int s  = m & (SEQ - 1);
        Vt[((size_t)(bb * KV + n)) * SEQ + s] = (__bf16)acc[mt][nt][r];
      }
}

// ---------------------------------------------------------------------------
// FC GEMM: out[M,2048](f32) = Ao[M,2048](bf16) * Wfc[2048,2048](f32)^T + bias
// ---------------------------------------------------------------------------
__global__ __launch_bounds__(128) void gemm_bf16_f32(
    const __bf16* __restrict__ X, const float* __restrict__ W,
    const float* __restrict__ bias, float* __restrict__ C) {
  const int lane = threadIdx.x & 31, wave = threadIdx.x >> 5;
  const int l16 = lane & 15, half = lane >> 4;
  const int m0 = blockIdx.x * 64;
  const int n0 = blockIdx.y * 256 + wave * 64;

  v8f acc[4][4];
#pragma unroll
  for (int i = 0; i < 4; ++i)
#pragma unroll
    for (int j = 0; j < 4; ++j) acc[i][j] = (v8f){0,0,0,0,0,0,0,0};

  for (int kc = 0; kc < EMBED; kc += 32) {
    __builtin_prefetch(X + (size_t)(m0 + l16) * EMBED + kc + 64, 0, 0);
    __builtin_prefetch(W + (size_t)(n0 + l16) * EMBED + kc + 64, 0, 0);
    v16bf aF[4], bF[4];
#pragma unroll
    for (int mt = 0; mt < 4; ++mt)
      aF[mt] = loadA_bf16(X + (size_t)(m0 + mt * 16 + l16) * EMBED + kc, half);
#pragma unroll
    for (int nt = 0; nt < 4; ++nt)
      bF[nt] = loadB_f32(W + (size_t)(n0 + nt * 16 + l16) * EMBED + kc, half);
#pragma unroll
    for (int mt = 0; mt < 4; ++mt)
#pragma unroll
      for (int nt = 0; nt < 4; ++nt)
        acc[mt][nt] = wmma_bf16(aF[mt], bF[nt], acc[mt][nt]);
  }

#pragma unroll
  for (int nt = 0; nt < 4; ++nt) {
    float bv = bias[n0 + nt * 16 + l16];
#pragma unroll
    for (int mt = 0; mt < 4; ++mt)
#pragma unroll
      for (int r = 0; r < 8; ++r) {
        int m = m0 + mt * 16 + r + 8 * half;
        int n = n0 + nt * 16 + l16;
        C[(size_t)m * EMBED + n] = acc[mt][nt][r] + bv;
      }
  }
}

// ---------------------------------------------------------------------------
// Flash GQA attention: one wave per (b, h, 16-query tile).
// Qp:[B*S,2048] Kp:[B*S,512] bf16 (row-major), Vt:[B*KV,S] bf16 (transposed),
// Ao:[B*S,2048] bf16
// ---------------------------------------------------------------------------
__global__ __launch_bounds__(128) void gqa_attn(
    const __bf16* __restrict__ Qp, const __bf16* __restrict__ Kp,
    const __bf16* __restrict__ Vt, __bf16* __restrict__ Ao) {
  __shared__ __bf16 sP[4][16 * 32];   // P tile (16q x 32k) per wave

  const int lane = threadIdx.x & 31, wave = threadIdx.x >> 5;
  const int l16 = lane & 15, half = lane >> 4;

  const int t  = blockIdx.x * 4 + wave;       // 0..8191
  const int qt = t & 63;
  const int h  = (t >> 6) & 31;
  const int b  = t >> 11;
  const int g  = h >> 2;

  const __bf16* Qbase = Qp + (size_t)(b * SEQ + qt * 16) * EMBED + h * HEAD_DIM;
  const __bf16* Kbase = Kp + (size_t)(b * SEQ) * KV + g * HEAD_DIM;
  const __bf16* Vtb   = Vt + (size_t)(b * KV + g * HEAD_DIM) * SEQ;

  // Q A-fragments for d=0..31 and d=32..63 (held for the whole pass)
  v16bf qA[2];
  {
    const __bf16* qr = Qbase + (size_t)l16 * EMBED;
#pragma unroll
    for (int c = 0; c < 2; ++c) {
      v8bf lo = *(const v8bf*)(qr + c * 32 + half * 16);
      v8bf hi = *(const v8bf*)(qr + c * 32 + half * 16 + 8);
      qA[c] = combine16(lo, hi);
    }
  }

  float runMax[8], runSum[8];
  v8f acc[4];
#pragma unroll
  for (int r = 0; r < 8; ++r) { runMax[r] = -1e30f; runSum[r] = 0.0f; }
#pragma unroll
  for (int dc = 0; dc < 4; ++dc) acc[dc] = (v8f){0,0,0,0,0,0,0,0};

  __bf16* pP = sP[wave];

  for (int nb = 0; nb < SEQ; nb += 32) {
    // scores: two 16x16 tiles over key columns nb..nb+31
    v8f st[2];
#pragma unroll
    for (int tk = 0; tk < 2; ++tk) {
      const __bf16* kr = Kbase + (size_t)(nb + tk * 16 + l16) * KV;
      v8f s = (v8f){0,0,0,0,0,0,0,0};
#pragma unroll
      for (int c = 0; c < 2; ++c) {
        v16bf kB = *(const v16bf*)(kr + c * 32 + half * 16);
        s = wmma_bf16(qA[c], kB, s);
      }
#pragma unroll
      for (int r = 0; r < 8; ++r) s[r] *= 0.125f;  // 1/sqrt(64)
      st[tk] = s;
    }

    // online softmax over this 32-key block (rows live in 16-lane halves)
    float corr[8];
#pragma unroll
    for (int r = 0; r < 8; ++r) {
      float mx = fmaxf(st[0][r], st[1][r]);
#pragma unroll
      for (int off = 1; off < 16; off <<= 1) mx = fmaxf(mx, __shfl_xor(mx, off, 32));
      float nmax = fmaxf(runMax[r], mx);
      corr[r] = __expf(runMax[r] - nmax);
      float e0 = __expf(st[0][r] - nmax);
      float e1 = __expf(st[1][r] - nmax);
      st[0][r] = e0; st[1][r] = e1;
      float sum = e0 + e1;
#pragma unroll
      for (int off = 1; off < 16; off <<= 1) sum += __shfl_xor(sum, off, 32);
      runSum[r] = runSum[r] * corr[r] + sum;
      runMax[r] = nmax;
    }

    // re-layout P (C-tile) -> A-fragment through LDS
#pragma unroll
    for (int tk = 0; tk < 2; ++tk)
#pragma unroll
      for (int r = 0; r < 8; ++r)
        pP[(r + 8 * half) * 32 + tk * 16 + l16] = (__bf16)st[tk][r];

    v16bf pA;
    {
      const __bf16* pr = pP + l16 * 32;
      v8bf lo = *(const v8bf*)(pr + half * 16);
      v8bf hi = *(const v8bf*)(pr + half * 16 + 8);
      pA = combine16(lo, hi);
    }

    // O = O*corr + P @ V   (B-fragment: 16 consecutive keys per lane, direct
    // contiguous 32B load from transposed Vt — no LDS staging needed)
#pragma unroll
    for (int dc = 0; dc < 4; ++dc) {
#pragma unroll
      for (int r = 0; r < 8; ++r) acc[dc][r] *= corr[r];
      v16bf vB = *(const v16bf*)(Vtb + (size_t)(dc * 16 + l16) * SEQ + nb + half * 16);
      acc[dc] = wmma_bf16(pA, vB, acc[dc]);
    }
  }

  // epilogue: normalize and store
  float inv[8];
#pragma unroll
  for (int r = 0; r < 8; ++r) inv[r] = 1.0f / runSum[r];
  __bf16* orow = Ao + (size_t)(b * SEQ + qt * 16) * EMBED + h * HEAD_DIM;
#pragma unroll
  for (int dc = 0; dc < 4; ++dc)
#pragma unroll
    for (int r = 0; r < 8; ++r)
      orow[(size_t)(r + 8 * half) * EMBED + dc * 16 + l16] =
          (__bf16)(acc[dc][r] * inv[r]);
}

// ---------------------------------------------------------------------------
extern "C" void kernel_launch(void* const* d_in, const int* in_sizes, int n_in,
                              void* d_out, int out_size, void* d_ws, size_t ws_size,
                              hipStream_t stream) {
  const float* q   = (const float*)d_in[0];
  const float* k   = (const float*)d_in[1];
  const float* v   = (const float*)d_in[2];
  const float* Wq  = (const float*)d_in[3];
  const float* Wk  = (const float*)d_in[4];
  const float* Wv  = (const float*)d_in[5];
  const float* Wfc = (const float*)d_in[6];
  const float* bfc = (const float*)d_in[7];
  float* out = (float*)d_out;

  char* ws = (char*)d_ws;
  const size_t qp_bytes = (size_t)ROWS * EMBED * 2;  // 16 MB
  const size_t kv_bytes = (size_t)ROWS * KV * 2;     // 4 MB
  __bf16* Qp = (__bf16*)(ws);
  __bf16* Kp = (__bf16*)(ws + qp_bytes);
  __bf16* Vt = (__bf16*)(ws + qp_bytes + kv_bytes);
  __bf16* Ao = (__bf16*)(ws + qp_bytes + 2 * kv_bytes);

  dim3 blk(128);
  gemm_f32_bf16   <<<dim3(ROWS / 64, EMBED / 256), blk, 0, stream>>>(q, Wq, Qp, EMBED);
  gemm_f32_bf16   <<<dim3(ROWS / 64, KV / 256),    blk, 0, stream>>>(k, Wk, Kp, KV);
  gemm_f32_bf16_vT<<<dim3(ROWS / 64, KV / 256),    blk, 0, stream>>>(v, Wv, Vt);
  gqa_attn<<<dim3((BATCH * HEADS * (SEQ / 16)) / 4), blk, 0, stream>>>(Qp, Kp, Vt, Ao);
  gemm_bf16_f32   <<<dim3(ROWS / 64, EMBED / 256), blk, 0, stream>>>(Ao, Wfc, bfc, out);
}
